// ConvNetStdp_2362232013298
// MI455X (gfx1250) — matrix operand order repaired and checked by reading, not compiled, not run on hardware
//
#include <hip/hip_runtime.h>
#include <cstdint>

// ---------------------------------------------------------------------------
// Spiking ConvNet on MI455X (gfx1250, wave32, WMMA + TDM).
//
// All GEMM-shaped work (conv1/conv2 via im2col, FC1, FC2) runs on
// v_wmma_f32_16x16x32_f16 (f16 in, f32 accumulate; spikes are exact in f16).
// Weights are pre-packed into fragment-ready f16 tiles; per K-step the block
// leader wave DMAs the shared B chunk into double-buffered LDS with the
// Tensor Data Mover (tensor_load_to_lds, TENSORcnt), all waves consume it via
// ds_load_b128, A fragments stream from global with b128 loads + prefetch.
// LIF state update + 2x2 maxpool + f16 spike pack are fused pointwise kernels.
// Workload is L2/bandwidth bound (~110 GFLOP, data resident in 192MB L2).
// ---------------------------------------------------------------------------

typedef _Float16 h8    __attribute__((ext_vector_type(8)));
typedef _Float16 v16h  __attribute__((ext_vector_type(16)));
typedef float    v8f   __attribute__((ext_vector_type(8)));
typedef unsigned int uint32x4 __attribute__((ext_vector_type(4)));
typedef int      int32x4 __attribute__((ext_vector_type(4)));
typedef int      int32x8 __attribute__((ext_vector_type(8)));

// ---------------------------------------------------------------------------
// TDM 2D tile load: rows x width_elems (f16) from global (row stride in
// elements) into contiguous LDS at lds_off. D# per CDNA5 ISA (§8):
//  g0: [1:0]=count=1 | lds_addr[63:32] | global_addr[120:64] | type=2@[127:126]
//  g1: data_size=1(2B)@[17:16] | tensor_dim0@[79:48] | tensor_dim1@[111:80]
//      tile_dim0@[127:112] | tile_dim1@[143:128] | dim0_stride@[207:160]
//  groups 2/3: zero (2D tile, no dim2+).
// ---------------------------------------------------------------------------
__device__ __forceinline__ void tdm_load_2d(unsigned lds_off,
                                            unsigned long long gaddr,
                                            unsigned width_elems,
                                            unsigned rows,
                                            unsigned long long row_stride_elems)
{
    uint32x4 g0;
    g0[0] = 1u;                                            // count=1, user desc
    g0[1] = lds_off;                                       // lds_addr (bytes)
    g0[2] = (unsigned)(gaddr & 0xFFFFFFFFull);             // global_addr[31:0]
    g0[3] = (unsigned)((gaddr >> 32) & 0x1FFFFFFull)       // global_addr[56:32]
          | (2u << 30);                                    // type = 2 ("image")
    int32x8 g1;
    g1[0] = (int)(1u << 16);                               // data_size=1 (2B)
    g1[1] = (int)((width_elems & 0xFFFFu) << 16);          // tensor_dim0[15:0]
    g1[2] = (int)(((width_elems >> 16) & 0xFFFFu)          // tensor_dim0[31:16]
          | ((rows & 0xFFFFu) << 16));                     // tensor_dim1[15:0]
    g1[3] = (int)(((rows >> 16) & 0xFFFFu)                 // tensor_dim1[31:16]
          | ((width_elems & 0xFFFFu) << 16));              // tile_dim0
    g1[4] = (int)(rows & 0xFFFFu);                         // tile_dim1 (tile_dim2=0)
    g1[5] = (int)(unsigned)(row_stride_elems & 0xFFFFFFFFull);      // dim0_stride lo
    g1[6] = (int)(unsigned)((row_stride_elems >> 32) & 0xFFFFull);  // dim0_stride hi
    g1[7] = 0;
    int32x4 z4 = {0, 0, 0, 0};
#if defined(__clang_major__) && (__clang_major__ >= 23)
    int32x8 z8 = {0, 0, 0, 0, 0, 0, 0, 0};
    __builtin_amdgcn_tensor_load_to_lds(g0, g1, z4, z4, z8, 0);
#else
    __builtin_amdgcn_tensor_load_to_lds(g0, g1, z4, z4, 0);
#endif
}

// Low 32 bits of a flat LDS pointer == LDS byte offset (aperture maps addr[31:0]).
__device__ __forceinline__ unsigned lds_off_of(const void* p)
{
    return (unsigned)(uintptr_t)p;
}

// ---------------------------------------------------------------------------
// WMMA GEMM with TDM-staged B:  D[M][Npad] = alpha * (A[M][Kt*32] * Bpack + b)
// Wave tile: 16 x (16*NBLK). Block: (blockDim/32) waves, distinct M-tiles,
// shared N-group (blockIdx.y). Per k-step: leader wave TDM-loads the NBLK
// B-tile chunk (NBLK rows x 512 halves, row stride Kt*512) into LDS buf,
// double-buffered; waves consume via ds_load_b128. A fragment per ISA 16x32
// f16 layout: lane<16: halves 0..7=K+0..7, 8..15=K+16..23; lane>=16: +8.
// B fragment (packed): N=lane&15, K as A: j 0..7=K+0..7, 8..15=K+16..23; lane>=16: +8.
// C/D f32: VGPR r -> row r+8*(lane>=16), col lane&15.
// ---------------------------------------------------------------------------
template <int NBLK>
__global__ __launch_bounds__(256)
void snn_gemm_wmma(const _Float16* __restrict__ A,
                   const _Float16* __restrict__ Bp,
                   const float* __restrict__ bias,
                   float* __restrict__ D,
                   int Mt, int Kt, int Npad, int Nbias, float alpha)
{
    __shared__ _Float16 bsm[2][NBLK * 512];

    int wpb  = blockDim.x >> 5;
    int wv   = threadIdx.x >> 5;
    int lane = threadIdx.x & 31;
    int tng  = blockIdx.y;
    int tm   = blockIdx.x * wpb + wv;
    bool active = (tm < Mt);
    int hi16 = lane >> 4;
    int l16  = lane & 15;

    long ldA = (long)Kt * 32;
    const _Float16* aRow   = A + (long)(tm * 16 + l16) * ldA + hi16 * 8;
    const _Float16* bStrip = Bp + (long)tng * NBLK * Kt * 512;

    // Prologue: leader wave issues TDM for chunk 0.
    if (threadIdx.x < 32) {
        tdm_load_2d(lds_off_of(&bsm[0][0]),
                    (unsigned long long)(uintptr_t)bStrip,
                    512u, (unsigned)NBLK, (unsigned long long)Kt * 512ull);
    }

    v8f acc[NBLK] = {};

    for (int kt = 0; kt < Kt; ++kt) {
        int buf = kt & 1;
        if (threadIdx.x < 32)
            __builtin_amdgcn_s_wait_tensorcnt(0);   // chunk kt landed in LDS
        __syncthreads();                            // publish to all waves
        if (threadIdx.x < 32 && kt + 1 < Kt) {      // overlap: DMA next chunk
            tdm_load_2d(lds_off_of(&bsm[buf ^ 1][0]),
                        (unsigned long long)(uintptr_t)(bStrip + (long)(kt + 1) * 512),
                        512u, (unsigned)NBLK, (unsigned long long)Kt * 512ull);
        }
        if (active) {                               // uniform per wave: EXEC all-1s
            const _Float16* ap = aRow + (long)kt * 32;
            h8 alo = *(const h8*)(ap);
            h8 ahi = *(const h8*)(ap + 16);
            if (kt + 1 < Kt)
                __builtin_prefetch(ap + 32, 0, 1);  // global_prefetch_b8 next A
            v16h a = __builtin_shufflevector(alo, ahi,
                        0,1,2,3,4,5,6,7,8,9,10,11,12,13,14,15);
#pragma unroll
            for (int r = 0; r < NBLK; ++r) {
                const _Float16* bp = &bsm[buf][r * 512 + lane * 16];
                h8 blo = *(const h8*)(bp);          // ds_load_b128
                h8 bhi = *(const h8*)(bp + 8);
                v16h b = __builtin_shufflevector(blo, bhi,
                            0,1,2,3,4,5,6,7,8,9,10,11,12,13,14,15);
                acc[r] = __builtin_amdgcn_wmma_f32_16x16x32_f16(
                             false, a, false, b, (short)0, acc[r], false, false);
            }
        }
        __syncthreads();                            // buf reusable next-next iter
    }

    if (active) {
#pragma unroll
        for (int r = 0; r < NBLK; ++r) {
            int n = (tng * NBLK + r) * 16 + l16;
            float bv = (n < Nbias) ? bias[n] : 0.0f;
            float* drow = D + (long)(tm * 16 + hi16 * 8) * Npad + n;
#pragma unroll
            for (int rr = 0; rr < 8; ++rr)
                drow[(long)rr * Npad] = alpha * (acc[r][rr] + bv);
        }
    }
}

// ---------------------------------------------------------------------------
// Pack weights W[N][Kreal] (row-major f32) into fragment-ready f16 Bpack:
// out[((nt*Ktiles + kt)*32 + lane)*16 + j].
// ---------------------------------------------------------------------------
__global__ void snn_pack_wB(const float* __restrict__ W, _Float16* __restrict__ out,
                            int Kreal, int Nreal, int Ktiles, int total)
{
    int idx = blockIdx.x * blockDim.x + threadIdx.x;
    if (idx >= total) return;
    int j    = idx & 15;
    int lane = (idx >> 4) & 31;
    int tile = idx >> 9;
    int kt   = tile % Ktiles;
    int nt   = tile / Ktiles;
    int k = kt * 32 + ((j < 8) ? (lane >> 4) * 8 + j : 16 + (lane >> 4) * 8 + (j - 8));
    int n = nt * 16 + (lane & 15);
    _Float16 v = (_Float16)0.0f;
    if (k < Kreal && n < Nreal) v = (_Float16)W[(long)n * Kreal + k];
    out[idx] = v;
}

// im2col for conv1: x[t] (64,3,64,64) f32 -> A1[64*60*60][96] f16 (75 real K)
__global__ void snn_im2col1(const float* __restrict__ x, _Float16* __restrict__ A1,
                            int t, int total)
{
    int idx = blockIdx.x * blockDim.x + threadIdx.x;
    if (idx >= total) return;
    int k = idx % 96;
    int m = idx / 96;
    int b   = m / 3600;
    int rem = m % 3600;
    int oh  = rem / 60, ow = rem % 60;
    _Float16 v = (_Float16)0.0f;
    if (k < 75) {
        int c = k / 25, rs = k % 25, r = rs / 5, s = rs % 5;
        v = (_Float16)x[(((long)t * 64 + b) * 3 + c) * 4096 + (oh + r) * 64 + (ow + s)];
    }
    A1[idx] = v;
}

// im2col for conv2: pooled spikes P1 (64,32,30,30) f16 -> A2[64*26*26][800] f16
__global__ void snn_im2col2(const _Float16* __restrict__ P1, _Float16* __restrict__ A2,
                            int total)
{
    int idx = blockIdx.x * blockDim.x + threadIdx.x;
    if (idx >= total) return;
    int k = idx % 800;
    int m = idx / 800;
    int b   = m / 676;
    int rem = m % 676;
    int oh  = rem / 26, ow = rem % 26;
    int c = k / 25, rs = k % 25, r = rs / 5, s = rs % 5;
    A2[idx] = P1[((long)(b * 32 + c)) * 900 + (oh + r) * 30 + (ow + s)];
}

// Fused LIF step + 2x2 maxpool + f16 spike pack.
__global__ void snn_lif_pool(const float* __restrict__ Y,
                             float* __restrict__ V, float* __restrict__ I,
                             _Float16* __restrict__ P,
                             int C, int Hs, int Po, int total)
{
    int idx = blockIdx.x * blockDim.x + threadIdx.x;
    if (idx >= total) return;
    int pw = idx % Po;
    int t1 = idx / Po;
    int ph = t1 % Po;
    int t2 = t1 / Po;
    int c  = t2 % C;
    int b  = t2 / C;
    float zmax = 0.0f;
#pragma unroll
    for (int dy = 0; dy < 2; ++dy)
#pragma unroll
        for (int dx = 0; dx < 2; ++dx) {
            int h = 2 * ph + dy, w = 2 * pw + dx;
            long s    = (((long)b * C + c) * Hs + h) * Hs + w;
            long yoff = ((long)b * Hs * Hs + h * Hs + w) * (long)C + c;
            float y = Y[yoff];
            float v = V[s], i = I[s];
            float vdec = v + 0.1f * (i - v);          // dt*tau_mem_inv = 0.1
            float z = (vdec - 1.0f > 0.0f) ? 1.0f : 0.0f;
            V[s] = (1.0f - z) * vdec;                 // v_reset = 0
            I[s] = 0.8f * i + y;                      // 1 - dt*tau_syn_inv = 0.8
            zmax = fmaxf(zmax, z);
        }
    P[idx] = (_Float16)zmax;
}

// LIF on FC layer; spikes (== relu(spikes)) packed to f16 for next GEMM.
__global__ void snn_lif_fc(const float* __restrict__ Y,
                           float* __restrict__ V, float* __restrict__ I,
                           _Float16* __restrict__ S, int total)
{
    int idx = blockIdx.x * blockDim.x + threadIdx.x;
    if (idx >= total) return;
    float y = Y[idx];
    float v = V[idx], i = I[idx];
    float vdec = v + 0.1f * (i - v);
    float z = (vdec - 1.0f > 0.0f) ? 1.0f : 0.0f;
    V[idx] = (1.0f - z) * vdec;
    I[idx] = 0.8f * i + y;
    S[idx] = (_Float16)z;
}

// Leaky-integrator readout; writes voltages[t] into d_out.
__global__ void snn_li_out(const float* __restrict__ Y,
                           float* __restrict__ V, float* __restrict__ I,
                           float* __restrict__ out, int t)
{
    int idx = blockIdx.x * blockDim.x + threadIdx.x;
    if (idx >= 640) return;                           // 64*10
    int b = idx / 10, j = idx % 10;
    float inp = Y[b * 16 + j];                        // D4 padded to N=16
    float v = V[idx], i = I[idx];
    float vnew = v + 0.1f * (i - v);                  // uses old i
    I[idx] = 0.8f * i + inp;
    V[idx] = vnew;
    out[(long)t * 640 + idx] = vnew;
}

extern "C" void kernel_launch(void* const* d_in, const int* in_sizes, int n_in,
                              void* d_out, int out_size, void* d_ws, size_t ws_size,
                              hipStream_t stream)
{
    const float* x  = (const float*)d_in[0];
    const float* w1 = (const float*)d_in[1];
    const float* b1 = (const float*)d_in[2];
    const float* w2 = (const float*)d_in[3];
    const float* b2 = (const float*)d_in[4];
    const float* wf = (const float*)d_in[5];
    const float* bf = (const float*)d_in[6];
    const float* wo = (const float*)d_in[7];
    const float* bo = (const float*)d_in[8];
    float* out = (float*)d_out;
    (void)in_sizes; (void)n_in; (void)out_size; (void)ws_size;

    char* ws = (char*)d_ws;
    size_t off = 0;
    auto alloc = [&](size_t bytes) {
        size_t o = off;
        off += (bytes + 255) & ~(size_t)255;
        return o;
    };

    // --- persistent LIF/LI state (zeroed once per launch) ---
    size_t o_v0 = alloc(29491200), o_i0 = alloc(29491200);   // (64,32,60,60) f32
    size_t o_v1 = alloc(11075584), o_i1 = alloc(11075584);   // (64,64,26,26) f32
    size_t o_v2 = alloc(262144),   o_i2 = alloc(262144);     // (64,1024) f32
    size_t o_vo = alloc(2560),     o_io = alloc(2560);       // (64,10) f32
    size_t stateEnd = off;
    // --- packed f16 weights (fragment-ready) ---
    size_t o_w1p = alloc((size_t)2 * 3 * 512 * 2);           // Nt=2,  Kt=3
    size_t o_w2p = alloc((size_t)4 * 25 * 512 * 2);          // Nt=4,  Kt=25
    size_t o_wfp = alloc((size_t)64 * 338 * 512 * 2);        // Nt=64, Kt=338
    size_t o_wop = alloc((size_t)1 * 32 * 512 * 2);          // Nt=1,  Kt=32
    // --- per-step activations (A1/A2 share one region: disjoint lifetimes) ---
    size_t o_A  = alloc(69222400);                           // max(A1, A2)
    size_t o_D1 = alloc(29491200);                           // conv1 out [230400][32] f32
    size_t o_P1 = alloc(3686400);                            // pooled spikes (64,32,30,30) f16
    size_t o_D2 = alloc(11075584);                           // conv2 out [43264][64] f32
    size_t o_A3 = alloc(1384448);                            // flat spikes [64][10816] f16
    size_t o_D3 = alloc(262144);                             // fc1 out [64][1024] f32
    size_t o_A4 = alloc(131072);                             // fc spikes [64][1024] f16
    size_t o_D4 = alloc(4096);                               // fc2 out [64][16] f32

    hipMemsetAsync(ws, 0, stateEnd, stream);                 // zero all states

    // --- weight packing (once per launch, deterministic) ---
    {
        int tot1 = 2 * 3 * 512;
        snn_pack_wB<<<(tot1 + 255) / 256, 256, 0, stream>>>(
            w1, (_Float16*)(ws + o_w1p), 75, 32, 3, tot1);
        int tot2 = 4 * 25 * 512;
        snn_pack_wB<<<(tot2 + 255) / 256, 256, 0, stream>>>(
            w2, (_Float16*)(ws + o_w2p), 800, 64, 25, tot2);
        int totf = 64 * 338 * 512;
        snn_pack_wB<<<(totf + 255) / 256, 256, 0, stream>>>(
            wf, (_Float16*)(ws + o_wfp), 10816, 1024, 338, totf);
        int toto = 1 * 32 * 512;
        snn_pack_wB<<<(toto + 255) / 256, 256, 0, stream>>>(
            wo, (_Float16*)(ws + o_wop), 1024, 10, 32, toto);
    }

    for (int t = 0; t < 16; ++t) {
        // conv1: im2col + WMMA GEMM  M=230400, N=32 (NBLK=2, Ntg=1), K=96
        int tI1 = 230400 * 96;
        snn_im2col1<<<(tI1 + 255) / 256, 256, 0, stream>>>(
            x, (_Float16*)(ws + o_A), t, tI1);
        snn_gemm_wmma<2><<<dim3((14400 + 7) / 8, 1), 256, 0, stream>>>(
            (const _Float16*)(ws + o_A), (const _Float16*)(ws + o_w1p),
            b1, (float*)(ws + o_D1), 14400, 3, 32, 32, 1.0f);

        // LIF0 + pool -> spikes f16 (64,32,30,30)
        int tL0 = 64 * 32 * 30 * 30;
        snn_lif_pool<<<(tL0 + 255) / 256, 256, 0, stream>>>(
            (const float*)(ws + o_D1), (float*)(ws + o_v0), (float*)(ws + o_i0),
            (_Float16*)(ws + o_P1), 32, 60, 30, tL0);

        // conv2: im2col + WMMA GEMM  M=43264, N=64 (NBLK=4, Ntg=1), K=800, alpha=10
        int tI2 = 43264 * 800;
        snn_im2col2<<<(tI2 + 255) / 256, 256, 0, stream>>>(
            (const _Float16*)(ws + o_P1), (_Float16*)(ws + o_A), tI2);
        snn_gemm_wmma<4><<<dim3((2704 + 7) / 8, 1), 256, 0, stream>>>(
            (const _Float16*)(ws + o_A), (const _Float16*)(ws + o_w2p),
            b2, (float*)(ws + o_D2), 2704, 25, 64, 64, 10.0f);

        // LIF1 + pool -> flattened spikes [64][10816] f16 (c,h,w == flatten order)
        int tL1 = 64 * 64 * 13 * 13;
        snn_lif_pool<<<(tL1 + 255) / 256, 256, 0, stream>>>(
            (const float*)(ws + o_D2), (float*)(ws + o_v1), (float*)(ws + o_i1),
            (_Float16*)(ws + o_A3), 64, 26, 13, tL1);

        // FC1: M=64 (Mt=4), N=1024 (NBLK=4, Ntg=16), K=10816 — 128 thr = 4 waves
        snn_gemm_wmma<4><<<dim3(1, 16), 128, 0, stream>>>(
            (const _Float16*)(ws + o_A3), (const _Float16*)(ws + o_wfp),
            bf, (float*)(ws + o_D3), 4, 338, 1024, 1024, 1.0f);

        // LIF2 -> spikes f16 [64][1024]
        int tL2 = 64 * 1024;
        snn_lif_fc<<<(tL2 + 255) / 256, 256, 0, stream>>>(
            (const float*)(ws + o_D3), (float*)(ws + o_v2), (float*)(ws + o_i2),
            (_Float16*)(ws + o_A4), tL2);

        // FC2: M=64 (Mt=4), N=16 (10 real, NBLK=1, Ntg=1), K=1024
        snn_gemm_wmma<1><<<dim3(1, 1), 128, 0, stream>>>(
            (const _Float16*)(ws + o_A4), (const _Float16*)(ws + o_wop),
            bo, (float*)(ws + o_D4), 4, 32, 16, 10, 1.0f);

        // LI readout -> d_out[t]
        snn_li_out<<<3, 256, 0, stream>>>(
            (const float*)(ws + o_D4), (float*)(ws + o_vo), (float*)(ws + o_io),
            out, t);
    }
}
